// GATReal_17222818857483
// MI455X (gfx1250) — matrix operand order repaired and checked by reading, not compiled, not run on hardware
//
#include <hip/hip_runtime.h>

typedef __bf16 bf16_t;
typedef bf16_t bf16x8  __attribute__((ext_vector_type(8)));
typedef bf16_t bf16x16 __attribute__((ext_vector_type(16)));
typedef float  floatx8 __attribute__((ext_vector_type(8)));

constexpr int NB  = 256;   // graphs
constexpr int KU  = 8;     // users / nodes per graph
constexpr int NTA = 64;    // antennas
constexpr int NH  = 40;    // attention heads
constexpr int NN  = NB * KU;        // 2048 nodes
constexpr int NE  = NN * KU;        // 16384 edges

__device__ __forceinline__ float wave_sum(float v) {
#pragma unroll
  for (int o = 16; o > 0; o >>= 1) v += __shfl_xor(v, o, 32);
  return v;
}

// ---------------- conversion kernels ----------------

__global__ __launch_bounds__(256)
void k_f32_to_bf16(const float* __restrict__ s, bf16_t* __restrict__ d, int n) {
  int i = blockIdx.x * 256 + threadIdx.x;
  if (i < n) d[i] = (bf16_t)s[i];
}

// src: Kd x Nc fp32 (row-major)  ->  dst: Nc x Kd bf16 (row-major)  (i.e. B^T)
__global__ __launch_bounds__(256)
void k_transpose_bf16(const float* __restrict__ src, bf16_t* __restrict__ dst,
                      int Kd, int Nc) {
  __shared__ float tile[32][33];
  int tx = threadIdx.x & 31, ty = threadIdx.x >> 5;
  int kb = blockIdx.y * 32, nb = blockIdx.x * 32;
#pragma unroll
  for (int i = 0; i < 32; i += 8) {
    int k = kb + ty + i, n = nb + tx;
    tile[ty + i][tx] = (k < Kd && n < Nc) ? src[(size_t)k * Nc + n] : 0.f;
  }
  __syncthreads();
#pragma unroll
  for (int i = 0; i < 32; i += 8) {
    int n = nb + ty + i, k = kb + tx;
    if (n < Nc && k < Kd) dst[(size_t)n * Kd + k] = (bf16_t)tile[tx][ty + i];
  }
}

// ---------------- WMMA bf16 GEMM (software-pipelined) ----------------

// A fragment: lanes 0-15 -> K[0..15]; lanes 16-31 -> K[16..31]
__device__ __forceinline__ bf16x16 ld_a(const bf16_t* p) {
  bf16x8 lo = *(const bf16x8*)p;
  bf16x8 hi = *(const bf16x8*)(p + 8);
  return __builtin_shufflevector(lo, hi, 0,1,2,3,4,5,6,7,8,9,10,11,12,13,14,15);
}
// B fragment: lane = column; lanes 0-15 K[0..15], lanes 16-31 K[16..31] (contiguous in Bt)
__device__ __forceinline__ bf16x16 ld_b(const bf16_t* p) {
  bf16x8 lo = *(const bf16x8*)p;
  bf16x8 hi = *(const bf16x8*)(p + 8);
  return __builtin_shufflevector(lo, hi, 0,1,2,3,4,5,6,7,8,9,10,11,12,13,14,15);
}
__device__ __forceinline__ floatx8 wmma_bf16(bf16x16 a, bf16x16 b, floatx8 c) {
  return __builtin_amdgcn_wmma_f32_16x16x32_bf16(false, a, false, b, (short)0, c,
                                                 false, false);
}

// C[M=2048][Nc] = A[M][Kd](bf16) * Bt[Nc][Kd]^T (bf16) + bias
// Block: 256 thr = 8 waves; tile 128(M) x 64(N); wave: 16 rows x 4 wmma n-tiles.
// K loop is register double-buffered so next iteration's 10 b128 loads are in
// flight while the current 4 WMMAs execute (avoids s_wait_loadcnt 0 per WMMA).
__global__ __launch_bounds__(256)
void k_gemm(const bf16_t* __restrict__ A, const bf16_t* __restrict__ Bt,
            const float* __restrict__ bias, float* __restrict__ C,
            int Nc, int Kd) {
  const int lane = threadIdx.x & 31;
  const int wave = threadIdx.x >> 5;
  const int m0   = blockIdx.y * 128 + wave * 16;
  const int n0   = blockIdx.x * 64;
  const int l15  = lane & 15;
  const bool hi  = lane >= 16;

  const bf16_t* Arow = A  + (size_t)(m0 + l15) * Kd + (hi ? 16 : 0);
  const bf16_t* B0   = Bt + (size_t)(n0 + l15) * Kd + (hi ? 16 : 0);
  const bf16_t* B1   = B0 + (size_t)16 * Kd;
  const bf16_t* B2   = B1 + (size_t)16 * Kd;
  const bf16_t* B3   = B2 + (size_t)16 * Kd;

  floatx8 acc0 = {}, acc1 = {}, acc2 = {}, acc3 = {};

  // prologue: fragments for k = 0
  bf16x16 a  = ld_a(Arow);
  bf16x16 b0 = ld_b(B0), b1 = ld_b(B1), b2 = ld_b(B2), b3 = ld_b(B3);

  for (int k = 32; k < Kd; k += 32) {
    // prefetch next iteration into independent registers
    bf16x16 an  = ld_a(Arow + k);
    bf16x16 bn0 = ld_b(B0 + k);
    bf16x16 bn1 = ld_b(B1 + k);
    bf16x16 bn2 = ld_b(B2 + k);
    bf16x16 bn3 = ld_b(B3 + k);
    // consume current fragments
    acc0 = wmma_bf16(a, b0, acc0);
    acc1 = wmma_bf16(a, b1, acc1);
    acc2 = wmma_bf16(a, b2, acc2);
    acc3 = wmma_bf16(a, b3, acc3);
    // rotate
    a = an; b0 = bn0; b1 = bn1; b2 = bn2; b3 = bn3;
  }
  acc0 = wmma_bf16(a, b0, acc0);
  acc1 = wmma_bf16(a, b1, acc1);
  acc2 = wmma_bf16(a, b2, acc2);
  acc3 = wmma_bf16(a, b3, acc3);

  // C/D layout: VGPR r holds row (m0+r) for lanes 0-15, (m0+8+r) for lanes 16-31
  const int rbase = m0 + (hi ? 8 : 0);
  {
    int cn = n0 + l15;      float bv = bias ? bias[cn] : 0.f;
#pragma unroll
    for (int r = 0; r < 8; ++r) C[(size_t)(rbase + r) * Nc + cn] = acc0[r] + bv;
  }
  {
    int cn = n0 + 16 + l15; float bv = bias ? bias[cn] : 0.f;
#pragma unroll
    for (int r = 0; r < 8; ++r) C[(size_t)(rbase + r) * Nc + cn] = acc1[r] + bv;
  }
  {
    int cn = n0 + 32 + l15; float bv = bias ? bias[cn] : 0.f;
#pragma unroll
    for (int r = 0; r < 8; ++r) C[(size_t)(rbase + r) * Nc + cn] = acc2[r] + bv;
  }
  {
    int cn = n0 + 48 + l15; float bv = bias ? bias[cn] : 0.f;
#pragma unroll
    for (int r = 0; r < 8; ++r) C[(size_t)(rbase + r) * Nc + cn] = acc3[r] + bv;
  }
}

// ---------------- GATv2 edge / softmax / aggregate ----------------

// one wave per (edge, head); ea@We computed on the fly (never materialize E x hc)
__global__ __launch_bounds__(256)
void k_edge_logits(const float* __restrict__ XL, const float* __restrict__ XR,
                   const float* __restrict__ ea, const float* __restrict__ We,
                   const float* __restrict__ att, const int* __restrict__ src,
                   const int* __restrict__ dst, float* __restrict__ logits,
                   int HC, int C) {
  int gw = blockIdx.x * 8 + (threadIdx.x >> 5);
  if (gw >= NE * NH) return;
  int lane = threadIdx.x & 31;
  int e = gw / NH, h = gw % NH;
  int s = src[e], t = dst[e];
  float ev0 = ea[e*6+0], ev1 = ea[e*6+1], ev2 = ea[e*6+2];
  float ev3 = ea[e*6+3], ev4 = ea[e*6+4], ev5 = ea[e*6+5];
  float acc = 0.f;
  for (int c = lane; c < C; c += 32) {
    int idx = h * C + c;
    float eev = ev0*We[0*HC+idx] + ev1*We[1*HC+idx] + ev2*We[2*HC+idx]
              + ev3*We[3*HC+idx] + ev4*We[4*HC+idx] + ev5*We[5*HC+idx];
    float z = XL[(size_t)s*HC + idx] + XR[(size_t)t*HC + idx] + eev;
    z = fmaxf(z, 0.f);                 // leaky_relu slope 0 -> relu
    acc += z * att[idx];
  }
  acc = wave_sum(acc);
  if (lane == 0) logits[(size_t)e * NH + h] = acc;
}

// softmax over the 8 incoming edges of each (node, head); e = b*64 + j*8 + i
__global__ __launch_bounds__(256)
void k_alpha(float* __restrict__ logits) {
  int tid = blockIdx.x * 256 + threadIdx.x;
  if (tid >= NN * NH) return;
  int n = tid / NH, h = tid % NH;
  int b = n >> 3, i = n & 7;
  int base = (b * 64 + i) * NH + h;    // + j*8*NH per source j
  float v[8]; float m = -1e30f;
#pragma unroll
  for (int j = 0; j < 8; ++j) { v[j] = logits[base + j * 8 * NH]; m = fmaxf(m, v[j]); }
  float ssum = 0.f;
#pragma unroll
  for (int j = 0; j < 8; ++j) { v[j] = expf(v[j] - m); ssum += v[j]; }
  float inv = 1.f / ssum;
#pragma unroll
  for (int j = 0; j < 8; ++j) logits[base + j * 8 * NH] = v[j] * inv;
}

// O[n][ch] (= x@Wres + bias, from GEMM) += sum_j alpha * XL[src_j][ch]
__global__ __launch_bounds__(256)
void k_aggregate(float* __restrict__ O, const float* __restrict__ XL,
                 const float* __restrict__ alpha, int HC, int C) {
  long tid = (long)blockIdx.x * 256 + threadIdx.x;
  if (tid >= (long)NN * HC) return;
  int n = (int)(tid / HC), ch = (int)(tid % HC);
  int h = ch / C;
  int b = n >> 3, i = n & 7;
  float acc = O[tid];
#pragma unroll
  for (int j = 0; j < 8; ++j)
    acc += alpha[(size_t)(b*64 + j*8 + i) * NH + h] * XL[(size_t)(b*8 + j) * HC + ch];
  O[tid] = acc;
}

// GraphNorm (stats over the 8 nodes of a graph, per channel) + ReLU, in place
__global__ __launch_bounds__(256)
void k_gnorm_relu(float* __restrict__ X, const float* __restrict__ w,
                  const float* __restrict__ bias, const float* __restrict__ ms, int D) {
  int tid = blockIdx.x * 256 + threadIdx.x;
  if (tid >= NB * D) return;
  int ch = tid % D, b = tid / D;
  float v[8]; float mean = 0.f;
#pragma unroll
  for (int j = 0; j < 8; ++j) { v[j] = X[(size_t)(b*8+j)*D + ch]; mean += v[j]; }
  mean *= 0.125f;
  float sub = ms[ch] * mean, var = 0.f;
#pragma unroll
  for (int j = 0; j < 8; ++j) { v[j] -= sub; var += v[j] * v[j]; }
  var *= 0.125f;
  float sc = w[ch] * rsqrtf(var + 1e-5f);
  float bb = bias[ch];
#pragma unroll
  for (int j = 0; j < 8; ++j)
    X[(size_t)(b*8+j)*D + ch] = fmaxf(sc * v[j] + bb, 0.f);
}

// ---------------- output heads ----------------

__global__ __launch_bounds__(256)
void k_rf_norm(const float* __restrict__ rf, float* __restrict__ out) {
  int tid = blockIdx.x * 256 + threadIdx.x;          // NN * NTA
  if (tid >= NN * NTA) return;
  int n = tid >> 6, t = tid & 63;
  float r  = rf[(size_t)n * 128 + t];
  float im = rf[(size_t)n * 128 + 64 + t];
  float mag = sqrtf(r * r + im * im) + 1e-12f;
  out[(size_t)tid * 2]     = r / mag;
  out[(size_t)tid * 2 + 1] = im / mag;
}

// one wave per edge: [x[src](512), ea(6), x[dst](512)] @ W[1030x2] + b
__global__ __launch_bounds__(256)
void k_bbout(const float* __restrict__ X, const float* __restrict__ ea,
             const float* __restrict__ W, const float* __restrict__ bvec,
             const int* __restrict__ src, const int* __restrict__ dst,
             float* __restrict__ bb) {
  int e = blockIdx.x * 8 + (threadIdx.x >> 5);
  if (e >= NE) return;
  int lane = threadIdx.x & 31;
  int s = src[e], t = dst[e];
  float a0 = 0.f, a1 = 0.f;
  for (int d = lane; d < 512; d += 32) {
    float xv = X[(size_t)s * 512 + d];
    a0 += xv * W[d * 2]; a1 += xv * W[d * 2 + 1];
  }
  if (lane < 6) {
    float v = ea[e * 6 + lane];
    a0 += v * W[(512 + lane) * 2]; a1 += v * W[(512 + lane) * 2 + 1];
  }
  for (int d = lane; d < 512; d += 32) {
    float xv = X[(size_t)t * 512 + d];
    a0 += xv * W[(518 + d) * 2]; a1 += xv * W[(518 + d) * 2 + 1];
  }
  a0 = wave_sum(a0); a1 = wave_sum(a1);
  if (lane == 0) { bb[e * 2] = a0 + bvec[0]; bb[e * 2 + 1] = a1 + bvec[1]; }
}

// Frobenius-normalize per graph (64 edges x 2 comps = 128 values); one wave per graph
__global__ __launch_bounds__(256)
void k_bb_norm(const float* __restrict__ bb, float* __restrict__ out) {
  int b = blockIdx.x * 8 + (threadIdx.x >> 5);
  if (b >= NB) return;
  int lane = threadIdx.x & 31;
  float ss = 0.f;
  for (int i = lane; i < 128; i += 32) { float v = bb[b * 128 + i]; ss += v * v; }
  ss = wave_sum(ss);
  float inv = 1.f / (sqrtf(ss) + 1e-12f);
  for (int i = lane; i < 128; i += 32) out[b * 128 + i] = bb[b * 128 + i] * inv;
}

__global__ __launch_bounds__(256)
void k_pout(const float* __restrict__ X, const float* __restrict__ W,
            const float* __restrict__ bv, float* __restrict__ pw) {
  int n = blockIdx.x * 8 + (threadIdx.x >> 5);
  if (n >= NN) return;
  int lane = threadIdx.x & 31;
  float acc = 0.f;
  for (int d = lane; d < 512; d += 32) acc += X[(size_t)n * 512 + d] * W[d];
  acc = wave_sum(acc);
  if (lane == 0) pw[n] = acc + bv[0];
}

__global__ __launch_bounds__(256)
void k_psoftmax(const float* __restrict__ pw, float* __restrict__ out) {
  int b = threadIdx.x;
  if (b >= NB) return;
  float v[8]; float m = -1e30f;
#pragma unroll
  for (int j = 0; j < 8; ++j) { v[j] = pw[b * 8 + j]; m = fmaxf(m, v[j]); }
  float s = 0.f;
#pragma unroll
  for (int j = 0; j < 8; ++j) { v[j] = expf(v[j] - m); s += v[j]; }
  float inv = 1.f / s;                 // p_max = 1.0
#pragma unroll
  for (int j = 0; j < 8; ++j) out[b * 8 + j] = v[j] * inv;
}

// ---------------- host side ----------------
// d_in layout (setup_inputs() dict order, params flattened leaf-by-leaf):
//  0 x, 1 edge_index, 2 edge_attr,
//  3..10  gat1 {Wl,bl,Wr,br,We,att,Wres,bias}, 11..18 gat2, 19..26 gat3,
//  27..29 gbn1 {w,b,ms}, 30..32 gbn2, 33..35 gbn3, 36..38 bn1, 39..41 bn2,
//  42,43 lin1 {W,b}, 44,45 lin2, 46,47 RFOut, 48,49 BBOut, 50,51 POut
extern "C" void kernel_launch(void* const* d_in, const int* in_sizes, int n_in,
                              void* d_out, int out_size, void* d_ws, size_t ws_size,
                              hipStream_t stream) {
  (void)in_sizes; (void)n_in; (void)out_size; (void)ws_size;
  auto F = [&](int i) -> const float* { return (const float*)d_in[i]; };
  const int* ei  = (const int*)d_in[1];
  const int* src = ei;
  const int* dst = ei + NE;
  const float* ea = F(2);

  const int dinA[3] = {128, 1280, 2560};
  const int hcA[3]  = {1280, 2560, 5120};
  const int cA[3]   = {32, 64, 128};
  const int gatI[3] = {3, 11, 19};
  const int gbnI[3] = {27, 30, 33};

  // ---- carve workspace ----
  size_t off = 0;
  auto carve = [&](size_t bytes) -> void* {
    off = (off + 255) & ~(size_t)255;
    void* p = (char*)d_ws + off;
    off += bytes;
    return p;
  };
  bf16_t* Wt[3][3];
  for (int l = 0; l < 3; ++l)
    for (int j = 0; j < 3; ++j)
      Wt[l][j] = (bf16_t*)carve((size_t)2 * hcA[l] * dinA[l]);
  bf16_t* lin1t = (bf16_t*)carve((size_t)2 * 1024 * 5120);
  bf16_t* lin2t = (bf16_t*)carve((size_t)2 * 512 * 1024);
  bf16_t* rft   = (bf16_t*)carve((size_t)2 * 128 * 512);
  bf16_t* Abf   = (bf16_t*)carve((size_t)2 * NN * 5120);
  float* O   = (float*)carve((size_t)4 * NN * 5120);
  float* XL  = (float*)carve((size_t)4 * NN * 5120);
  float* XR  = (float*)carve((size_t)4 * NN * 5120);
  float* LOG = (float*)carve((size_t)4 * NE * NH);
  float* RF  = (float*)carve((size_t)4 * NN * 128);
  float* BBv = (float*)carve((size_t)4 * NE * 2);
  float* PW  = (float*)carve((size_t)4 * NN);

  // ---- weight transpose+convert to bf16 (Bt layout: Nc x Kd) ----
  auto transpose = [&](const float* s, bf16_t* d, int Kd, int Nc) {
    dim3 g((Nc + 31) / 32, (Kd + 31) / 32);
    k_transpose_bf16<<<g, 256, 0, stream>>>(s, d, Kd, Nc);
  };
  for (int l = 0; l < 3; ++l) {
    transpose(F(gatI[l] + 0), Wt[l][0], dinA[l], hcA[l]);  // Wl
    transpose(F(gatI[l] + 2), Wt[l][1], dinA[l], hcA[l]);  // Wr
    transpose(F(gatI[l] + 6), Wt[l][2], dinA[l], hcA[l]);  // Wres
  }
  transpose(F(42), lin1t, 5120, 1024);
  transpose(F(44), lin2t, 1024, 512);
  transpose(F(46), rft, 512, 128);

  auto gemm = [&](const bf16_t* A, const bf16_t* Bt, const float* bias,
                  float* Cc, int Nc, int Kd) {
    dim3 g(Nc / 64, NN / 128);
    k_gemm<<<g, 256, 0, stream>>>(A, Bt, bias, Cc, Nc, Kd);
  };

  // ---- GAT stack ----
  k_f32_to_bf16<<<(NN * 128) / 256, 256, 0, stream>>>(F(0), Abf, NN * 128);
  for (int l = 0; l < 3; ++l) {
    int din = dinA[l], hc = hcA[l], C = cA[l];
    gemm(Abf, Wt[l][0], F(gatI[l] + 1), XL, hc, din);                  // xl = x@Wl + bl
    gemm(Abf, Wt[l][1], F(gatI[l] + 3), XR, hc, din);                  // xr = x@Wr + br
    gemm(Abf, Wt[l][2], F(gatI[l] + 7), O,  hc, din);                  // res = x@Wres + bias
    k_edge_logits<<<(NE * NH) / 8, 256, 0, stream>>>(
        XL, XR, ea, F(gatI[l] + 4), F(gatI[l] + 5), src, dst, LOG, hc, C);
    k_alpha<<<(NN * NH) / 256, 256, 0, stream>>>(LOG);
    k_aggregate<<<(NN * hc) / 256, 256, 0, stream>>>(O, XL, LOG, hc, C);
    k_gnorm_relu<<<(NB * hc) / 256, 256, 0, stream>>>(
        O, F(gbnI[l] + 0), F(gbnI[l] + 1), F(gbnI[l] + 2), hc);
    k_f32_to_bf16<<<(NN * hc) / 256, 256, 0, stream>>>(O, Abf, NN * hc);
  }

  // ---- MLP ----
  gemm(Abf, lin1t, F(43), O, 1024, 5120);
  k_gnorm_relu<<<(NB * 1024) / 256, 256, 0, stream>>>(O, F(36), F(37), F(38), 1024);
  k_f32_to_bf16<<<(NN * 1024) / 256, 256, 0, stream>>>(O, Abf, NN * 1024);
  gemm(Abf, lin2t, F(45), XL, 512, 1024);                               // XL := final x (N x 512)
  k_gnorm_relu<<<(NB * 512) / 256, 256, 0, stream>>>(XL, F(39), F(40), F(41), 512);
  k_f32_to_bf16<<<(NN * 512) / 256, 256, 0, stream>>>(XL, Abf, NN * 512);

  // ---- heads ----
  float* out = (float*)d_out;
  gemm(Abf, rft, F(47), RF, 128, 512);
  k_rf_norm<<<(NN * NTA) / 256, 256, 0, stream>>>(RF, out);
  k_bbout<<<NE / 8, 256, 0, stream>>>(XL, ea, F(48), F(49), src, dst, BBv);
  k_bb_norm<<<NB / 8, 256, 0, stream>>>(BBv, out + (size_t)NN * NTA * 2);
  k_pout<<<NN / 8, 256, 0, stream>>>(XL, F(50), F(51), PW);
  k_psoftmax<<<1, 256, 0, stream>>>(PW, out + (size_t)NN * NTA * 2 + (size_t)NE * 2);
}
